// EncoderRNN_23922967838798
// MI455X (gfx1250) — compile-verified, matrix operand order checked by reading, not run on hardware
//
#include <hip/hip_runtime.h>
#include <hip/hip_bf16.h>
#include <math.h>

typedef __attribute__((ext_vector_type(16))) __bf16 v16bf;
typedef __attribute__((ext_vector_type(8)))  __bf16 v8bf;
typedef __attribute__((ext_vector_type(8)))  float  v8f;

namespace {
constexpr int  kH  = 1024;                 // hidden size
constexpr int  kB  = 64;                   // batch
constexpr int  kT  = 512;                  // sequence length
constexpr long kBH = (long)kB * kH;        // 65536
constexpr long kWE = (long)4 * kH * kH;    // elems per weight matrix (4Mi)
constexpr long kXE = (long)kT * kB * kH;   // gathered x elems (33.5Mi)
}

// ---------------------------------------------------------------------------
// WMMA helpers (gfx1250, wave32): v_wmma_f32_16x16x32_bf16.
// 16-bit A-matrix (16x32 MxK) lane layout (ISA 7.12.2):
//   lanes 0-15 : row=lane,    elems 0..7 -> K 0..7,   elems 8..15 -> K 16..23
//   lanes 16-31: row=lane-16, elems 0..7 -> K 8..15,  elems 8..15 -> K 24..31
// B (32x16 KxN) mirrors with N=lane&15; W is row-major (4H,H)=(n,k) so each
// lane reads contiguous K from its row -> two 16B global_load_b128 per tile.
// All loads below are  uniform-SGPR-base + loop-invariant-VGPR-offset + imm,
// so the K loop advances only scalar base pointers (SALU, no per-iter VALU).
// ---------------------------------------------------------------------------
__device__ __forceinline__ v16bf mk_tile(const __bf16* base, int off1, int off2) {
  const v8bf lo = *reinterpret_cast<const v8bf*>(base + off1);
  const v8bf hi = *reinterpret_cast<const v8bf*>(base + off2);
  v16bf r;
#pragma unroll
  for (int e = 0; e < 8; ++e) { r[e] = lo[e]; r[e + 8] = hi[e]; }
  return r;
}

__device__ __forceinline__ v8f wmma_bf16(v16bf a, v16bf b, v8f c) {
  return __builtin_amdgcn_wmma_f32_16x16x32_bf16(false, a, false, b, (short)0, c,
                                                 false, false);
}

__device__ __forceinline__ float sigmoidf_(float x) {
  return 1.0f / (1.0f + __expf(-x));
}

// ---------------------------------------------------------------------------
// Fused bidirectional LSTM step.  g = [x_t ; h] @ [W_ih ; W_hh]^T + b  (K=2H)
// grid = (64, 2): blockIdx.x = 16-wide hidden tile, blockIdx.y = direction.
// block = 128 threads = 4 waves; wave w owns GATE w and accumulates all four
// 16-row batch tiles against one B tile (4x less W traffic from L2).  Gates
// are recombined via LDS, then the LSTM cell is applied elementwise and both
// directions atomically accumulate into out (zero-initialized; f32 add is
// commutative, so 0+hf+hb is order-independent -> deterministic).
// ---------------------------------------------------------------------------
struct StepArgs {
  const __bf16* x;          // (T, B, H) bf16 gathered embeddings
  const __bf16* Wih[2];     // (4H, H) bf16 per direction
  const __bf16* Whh[2];
  const float*  bias[2];    // (4H)
  const __bf16* h_in[2];    // (B, H) bf16
  __bf16*       h_out[2];
  float*        c[2];       // (B, H) f32, in-place
  float*        hT[2];      // (B, H) f32 (last write = final hidden)
  float*        out;        // (T, B, H) f32, atomically accumulated
  int           s;          // step index 0..T-1
};

__global__ __launch_bounds__(128)
void lstm_step_kernel(StepArgs args) {
  const int lane = threadIdx.x & 31;
  const int g    = threadIdx.x >> 5;          // gate index (wave id)
  const int half = lane >> 4;
  const int lr   = lane & 15;
  const int n0   = blockIdx.x * 16;
  const int dir  = blockIdx.y;

  const int t_x = dir ? (kT - 1 - args.s) : args.s;

  // loop-invariant per-lane element offsets (fit 32-bit, saddr+voffset form)
  const int co1 = half ? 8 : 0;
  const int co2 = half ? 24 : 16;
  const int a1  = lr * kH + co1;               // A tiles: x rows / h rows
  const int a2  = lr * kH + co2;
  const int b1  = (n0 + lr) * kH + co1;        // B tile: W row (n, k)
  const int b2  = (n0 + lr) * kH + co2;

  v8f acc[4];                                  // one per batch tile (m0=mt*16)
#pragma unroll
  for (int mt = 0; mt < 4; ++mt)
#pragma unroll
    for (int r = 0; r < 8; ++r) acc[mt][r] = 0.0f;

  // ---- input contribution: A = x[t_x] rows, B = W_ih gate g ---------------
  {
    const __bf16* xk = args.x + (long)t_x * kBH;           // uniform base
    const __bf16* wk = args.Wih[dir] + (long)g * kH * kH;  // uniform base
    for (int k = 0; k < kH; k += 32, xk += 32, wk += 32) {
      const v16bf bt = mk_tile(wk, b1, b2);
#pragma unroll
      for (int mt = 0; mt < 4; ++mt)
        acc[mt] = wmma_bf16(mk_tile(xk + mt * (16 * kH), a1, a2), bt, acc[mt]);
    }
  }
  // ---- recurrent contribution: A = h rows, B = W_hh gate g ----------------
  {
    const __bf16* hk = args.h_in[dir];
    const __bf16* wk = args.Whh[dir] + (long)g * kH * kH;
    for (int k = 0; k < kH; k += 32, hk += 32, wk += 32) {
      const v16bf bt = mk_tile(wk, b1, b2);
#pragma unroll
      for (int mt = 0; mt < 4; ++mt)
        acc[mt] = wmma_bf16(mk_tile(hk + mt * (16 * kH), a1, a2), bt, acc[mt]);
    }
  }

  // ---- recombine gates through LDS ----------------------------------------
  // C/D layout: lanes 0-15 hold M=r, N=lane; lanes 16-31 hold M=8+r.
  __shared__ float smem[4 * kB * 16];          // [gate][m][nn]  16 KB
#pragma unroll
  for (int mt = 0; mt < 4; ++mt)
#pragma unroll
    for (int r = 0; r < 8; ++r)
      smem[g * (kB * 16) + (mt * 16 + half * 8 + r) * 16 + lr] = acc[mt][r];
  __syncthreads();

  // ---- LSTM cell: each thread owns 8 (m, n) cells -------------------------
  const int nn = threadIdx.x & 15;
  const int n  = n0 + nn;
  const float bi = args.bias[dir][0 * kH + n];
  const float bff = args.bias[dir][1 * kH + n];
  const float bgg = args.bias[dir][2 * kH + n];
  const float boo = args.bias[dir][3 * kH + n];
  float*       cs   = args.c[dir];
  __bf16*      hn   = args.h_out[dir];
  float*       hT   = args.hT[dir];
  float*       outt = args.out + (long)t_x * kBH;
  const int mbase = threadIdx.x >> 4;          // 0..7
#pragma unroll
  for (int j = 0; j < 8; ++j) {
    const int  m   = mbase + 8 * j;
    const int  l   = m * 16 + nn;              // == threadIdx.x + 128*j
    const long idx = (long)m * kH + n;
    const float si = sigmoidf_(smem[0 * (kB * 16) + l] + bi);
    const float sf = sigmoidf_(smem[1 * (kB * 16) + l] + bff);
    const float tg = tanhf(    smem[2 * (kB * 16) + l] + bgg);
    const float so = sigmoidf_(smem[3 * (kB * 16) + l] + boo);
    const float cn = sf * cs[idx] + si * tg;
    const float hv = so * tanhf(cn);
    cs[idx] = cn;
    hn[idx] = (__bf16)hv;
    hT[idx] = hv;
    atomicAdd(outt + idx, hv);
  }
}

// ---------------------------------------------------------------------------
// Prep / epilogue kernels
// ---------------------------------------------------------------------------
__global__ void cvt_f32_bf16_kernel(const float* __restrict__ src,
                                    __bf16* __restrict__ dst, long n) {
  long i = (long)blockIdx.x * blockDim.x + threadIdx.x;
  const long stride = (long)gridDim.x * blockDim.x;
  for (; i < n; i += stride) dst[i] = (__bf16)src[i];
}

// x[t,b,:] = bf16(emb[tokens[t,b], :]) — one block per (t,b) row.
__global__ void gather_x_kernel(const float* __restrict__ emb,
                                const int* __restrict__ tokens,
                                __bf16* __restrict__ x) {
  const long row = blockIdx.x;                 // 0 .. T*B-1
  const int  tok = tokens[row];
  const float* src = emb + (long)tok * kH;
  __bf16*      dst = x + row * kH;
  for (int i = threadIdx.x; i < kH; i += blockDim.x) dst[i] = (__bf16)src[i];
}

__global__ void zero_kernel(float* __restrict__ p, long n) {
  long i = (long)blockIdx.x * blockDim.x + threadIdx.x;
  const long stride = (long)gridDim.x * blockDim.x;
  for (; i < n; i += stride) p[i] = 0.0f;
}

__global__ void init_state_kernel(const float* __restrict__ hidden,
                                  __bf16* hf, __bf16* hb,
                                  float* cf, float* cb,
                                  float* hTf, float* hTb) {
  const long i = (long)blockIdx.x * blockDim.x + threadIdx.x;
  if (i < kBH) {
    const float vf = hidden[i];
    const float vb = hidden[kBH + i];
    hf[i] = (__bf16)vf;  hb[i] = (__bf16)vb;
    cf[i] = 0.0f;        cb[i] = 0.0f;
    hTf[i] = vf;         hTb[i] = vb;
  }
}

__global__ void finalize_kernel(const float* __restrict__ hTf,
                                const float* __restrict__ hTb,
                                float* __restrict__ out_hidden) {
  const long i = (long)blockIdx.x * blockDim.x + threadIdx.x;
  if (i < kBH) {
    out_hidden[i]       = hTf[i];
    out_hidden[kBH + i] = hTb[i];
  }
}

// ---------------------------------------------------------------------------
extern "C" void kernel_launch(void* const* d_in, const int* in_sizes, int n_in,
                              void* d_out, int out_size, void* d_ws, size_t ws_size,
                              hipStream_t stream) {
  const int*   tokens    = (const int*)d_in[0];   // (T, B)
  const float* hidden    = (const float*)d_in[1]; // (2, B, H)
  const float* emb       = (const float*)d_in[3]; // (V, H)
  const float* Wih_f_f32 = (const float*)d_in[4];
  const float* Whh_f_f32 = (const float*)d_in[5];
  const float* b_f       = (const float*)d_in[6];
  const float* Wih_b_f32 = (const float*)d_in[7];
  const float* Whh_b_f32 = (const float*)d_in[8];
  const float* b_b       = (const float*)d_in[9];
  float* out = (float*)d_out;

  // ---- workspace carve-up (~98 MB, L2-resident) ---------------------------
  char* ws = (char*)d_ws;
  __bf16* Wih[2]; __bf16* Whh[2];
  Wih[0] = (__bf16*)ws; ws += kWE * sizeof(__bf16);
  Whh[0] = (__bf16*)ws; ws += kWE * sizeof(__bf16);
  Wih[1] = (__bf16*)ws; ws += kWE * sizeof(__bf16);
  Whh[1] = (__bf16*)ws; ws += kWE * sizeof(__bf16);
  __bf16* xg = (__bf16*)ws; ws += kXE * sizeof(__bf16);
  __bf16* hbuf[2][2];
  for (int d = 0; d < 2; ++d)
    for (int p = 0; p < 2; ++p) { hbuf[d][p] = (__bf16*)ws; ws += kBH * sizeof(__bf16); }
  float* cst[2]; for (int d = 0; d < 2; ++d) { cst[d] = (float*)ws; ws += kBH * sizeof(float); }
  float* hTs[2]; for (int d = 0; d < 2; ++d) { hTs[d] = (float*)ws; ws += kBH * sizeof(float); }

  // ---- prep (all massively parallel, off the recurrence critical path) ----
  cvt_f32_bf16_kernel<<<1024, 256, 0, stream>>>(Wih_f_f32, Wih[0], kWE);
  cvt_f32_bf16_kernel<<<1024, 256, 0, stream>>>(Whh_f_f32, Whh[0], kWE);
  cvt_f32_bf16_kernel<<<1024, 256, 0, stream>>>(Wih_b_f32, Wih[1], kWE);
  cvt_f32_bf16_kernel<<<1024, 256, 0, stream>>>(Whh_b_f32, Whh[1], kWE);
  gather_x_kernel<<<kT * kB, 256, 0, stream>>>(emb, tokens, xg);
  zero_kernel<<<2048, 256, 0, stream>>>(out, (long)kT * kBH);
  init_state_kernel<<<(kBH + 255) / 256, 256, 0, stream>>>(
      hidden, hbuf[0][0], hbuf[1][0], cst[0], cst[1], hTs[0], hTs[1]);

  // ---- 512 fused bidirectional steps --------------------------------------
  StepArgs a;
  a.x = xg;
  for (int d = 0; d < 2; ++d) {
    a.Wih[d] = Wih[d]; a.Whh[d] = Whh[d];
    a.c[d] = cst[d];   a.hT[d] = hTs[d];
  }
  a.bias[0] = b_f; a.bias[1] = b_b;
  a.out = out;

  for (int s = 0; s < kT; ++s) {
    const int cur = s & 1, nxt = cur ^ 1;
    for (int d = 0; d < 2; ++d) { a.h_in[d] = hbuf[d][cur]; a.h_out[d] = hbuf[d][nxt]; }
    a.s = s;
    lstm_step_kernel<<<dim3(kH / 16, 2), 128, 0, stream>>>(a);
  }

  // ---- final hidden states -------------------------------------------------
  finalize_kernel<<<(kBH + 255) / 256, 256, 0, stream>>>(
      hTs[0], hTs[1], out + (long)kT * kBH);
}
